// ProteinGNNTransformer_56100862821029
// MI455X (gfx1250) — compile-verified, matrix operand order checked by reading, not run on hardware
//
#include <hip/hip_runtime.h>
#include <hip/hip_bf16.h>
#include <math.h>

// ---------------------------------------------------------------------------
// Types for CDNA5 WMMA
// ---------------------------------------------------------------------------
typedef __bf16 bf16x16 __attribute__((ext_vector_type(16)));
typedef __bf16 bf16x8  __attribute__((ext_vector_type(8)));
typedef __bf16 bf16x4  __attribute__((ext_vector_type(4)));
typedef float  f32x8   __attribute__((ext_vector_type(8)));

static __device__ __forceinline__ f32x8 wmma_bf16(bf16x16 a, bf16x16 b, f32x8 c) {
  return __builtin_amdgcn_wmma_f32_16x16x32_bf16(
      /*neg_a=*/false, a, /*neg_b=*/false, b,
      /*c_mod=*/(short)0, c, /*reuse_a=*/false, /*reuse_b=*/false);
}

// A-layout 16x32 fragment from row-major bf16 (ld in elements), base=(row0,k0).
// Lane layout: two contiguous 16B runs -> two b128 loads, no conversion math.
static __device__ __forceinline__ bf16x16
load_frag_row_bf(const __bf16* __restrict__ base, int ld, int lane) {
  const int m = lane & 15, half = lane >> 4;
  const __bf16* p = base + (size_t)m * (size_t)ld + half * 8;
  const bf16x8 lo = *(const bf16x8*)(p);
  const bf16x8 hi = *(const bf16x8*)(p + 16);
  bf16x16 r;
#pragma unroll
  for (int i = 0; i < 8; ++i) { r[i] = lo[i]; r[8 + i] = hi[i]; }
  return r;
}

// Pre-packed B fragment: tile stored lane-major, 16 bf16 (32B) per lane.
static __device__ __forceinline__ bf16x16
load_frag_packed(const __bf16* __restrict__ Wp, int tile, int lane) {
  return *(const bf16x16*)(Wp + (size_t)tile * 512 + (size_t)lane * 16);
}

// ---------------------------------------------------------------------------
// Pack weight W[K x Nc] (f32 row-major) into per-lane B-fragment order (bf16).
// Columns < scaleCols are pre-scaled (folds 1/sqrt(DH) into Q projection).
// ---------------------------------------------------------------------------
__global__ void __launch_bounds__(256)
pack_w_kernel(const float* __restrict__ W, __bf16* __restrict__ out,
              int K, int Nc, int scaleCols, float scale) {
  const int idx = blockIdx.x * 256 + threadIdx.x;
  const int tiles = (K >> 5) * (Nc >> 4);
  if (idx >= tiles * 32) return;
  const int tile = idx >> 5, lane = idx & 31;
  const int tn = Nc >> 4;
  const int k0 = (tile / tn) << 5, n0 = (tile % tn) << 4;
  const int n = lane & 15, half = lane >> 4;
  const float s = (n0 + n < scaleCols) ? scale : 1.0f;
  bf16x16 r;
#pragma unroll
  for (int i = 0; i < 8; ++i) {
    const int kb = ((i & 4) << 2) + half * 8 + ((i & 3) << 1);
    r[2 * i]     = (__bf16)(W[(size_t)(k0 + kb) * Nc + n0 + n] * s);
    r[2 * i + 1] = (__bf16)(W[(size_t)(k0 + kb + 1) * Nc + n0 + n] * s);
  }
  *(bf16x16*)(out + (size_t)idx * 16) = r;
}

// ---------------------------------------------------------------------------
// WMMA GEMM: C[M x Nc] = act(A[M x K] @ B + bias); A bf16 row-major,
// B pre-packed bf16 fragments. One wave -> 32x64 tile (2 A x 4 B -> 8 WMMA).
// ---------------------------------------------------------------------------
template <bool RELU, bool HASBIAS, bool OUTBF16>
__global__ void __launch_bounds__(256)
gemm_wmma(const __bf16* __restrict__ A, const __bf16* __restrict__ Wp,
          const float* __restrict__ bias, void* __restrict__ Cv,
          int M, int K, int Nc) {
  const int lane = threadIdx.x & 31;
  const int wave = threadIdx.x >> 5;
  const int tilesN = Nc >> 6;
  const int tilesNc16 = Nc >> 4;
  const int gw = blockIdx.x * 8 + wave;
  const int tm = gw / tilesN;
  const int tn = gw % tilesN;
  if (tm * 32 >= M) return;
  const int row0 = tm * 32, col0 = tn * 64;

  f32x8 acc[2][4] = {};
  for (int k0 = 0; k0 < K; k0 += 32) {
    const int tbase = (k0 >> 5) * tilesNc16 + (col0 >> 4);
    if (k0 + 32 < K) {  // gfx1250 global_prefetch of next K-step operands
      __builtin_prefetch(A + (size_t)(row0 + (lane & 15)) * K + k0 + 32, 0, 1);
      __builtin_prefetch(Wp + (size_t)(tbase + tilesNc16) * 512 + lane * 16, 0, 1);
    }
    bf16x16 a0 = load_frag_row_bf(A + (size_t)row0 * K + k0, K, lane);
    bf16x16 a1 = load_frag_row_bf(A + (size_t)(row0 + 16) * K + k0, K, lane);
    bf16x16 b0 = load_frag_packed(Wp, tbase + 0, lane);
    bf16x16 b1 = load_frag_packed(Wp, tbase + 1, lane);
    bf16x16 b2 = load_frag_packed(Wp, tbase + 2, lane);
    bf16x16 b3 = load_frag_packed(Wp, tbase + 3, lane);
    acc[0][0] = wmma_bf16(a0, b0, acc[0][0]);
    acc[0][1] = wmma_bf16(a0, b1, acc[0][1]);
    acc[0][2] = wmma_bf16(a0, b2, acc[0][2]);
    acc[0][3] = wmma_bf16(a0, b3, acc[0][3]);
    acc[1][0] = wmma_bf16(a1, b0, acc[1][0]);
    acc[1][1] = wmma_bf16(a1, b1, acc[1][1]);
    acc[1][2] = wmma_bf16(a1, b2, acc[1][2]);
    acc[1][3] = wmma_bf16(a1, b3, acc[1][3]);
  }

  const int n = lane & 15, half = lane >> 4;
#pragma unroll
  for (int r = 0; r < 2; ++r) {
#pragma unroll
    for (int v = 0; v < 8; ++v) {
      const int m = r * 16 + v + half * 8;
      float x0 = acc[r][0][v], x1 = acc[r][1][v];
      float x2 = acc[r][2][v], x3 = acc[r][3][v];
      if (HASBIAS) {
        x0 += bias[col0 + n];
        x1 += bias[col0 + 16 + n];
        x2 += bias[col0 + 32 + n];
        x3 += bias[col0 + 48 + n];
      }
      if (RELU) {
        x0 = fmaxf(x0, 0.f); x1 = fmaxf(x1, 0.f);
        x2 = fmaxf(x2, 0.f); x3 = fmaxf(x3, 0.f);
      }
      if (OUTBF16) {
        __bf16* crow = (__bf16*)Cv + (size_t)(row0 + m) * Nc + col0;
        crow[n] = (__bf16)x0; crow[16 + n] = (__bf16)x1;
        crow[32 + n] = (__bf16)x2; crow[48 + n] = (__bf16)x3;
      } else {
        float* crow = (float*)Cv + (size_t)(row0 + m) * Nc + col0;
        crow[n] = x0; crow[16 + n] = x1; crow[32 + n] = x2; crow[48 + n] = x3;
      }
    }
  }
}

// ---------------------------------------------------------------------------
// Flash attention: ONE WAVE PER WORKGROUP (barriers -> S_NOP on CDNA5),
// one wave per (head, 16-query tile), 32-key chunks. 1/sqrt(DH) already
// folded into the Q projection weights.
// qkv bf16 [4096 x 768]; vT bf16 [256 dims x 4096 keys]; o bf16 [4096 x 256]
// ---------------------------------------------------------------------------
__global__ void __launch_bounds__(32)
attention_kernel(const __bf16* __restrict__ qkv, const __bf16* __restrict__ vT,
                 __bf16* __restrict__ o) {
  __shared__ float lds[544];  // 16x32 score tile + 32 stats
  const int lane = threadIdx.x;
  float* S = lds;
  float* stats = lds + 512;

  const int gw = blockIdx.x;             // 0..2047
  const int h = gw >> 8;                 // head 0..7
  const int qt = gw & 255;               // query tile 0..255
  const int row0 = qt * 16;
  const int ld = 768;
  const __bf16* Qb = qkv + (size_t)row0 * ld + h * 32;
  const __bf16* Kb = qkv + 256 + h * 32;
  const __bf16* V0 = vT + (size_t)(h * 32) * 4096;
  const __bf16* V1 = vT + (size_t)(h * 32 + 16) * 4096;

  const int n = lane & 15, half = lane >> 4;
  bf16x16 qf = load_frag_row_bf(Qb, ld, lane);

  f32x8 o0 = {}, o1 = {};
  float run_max = -3.0e38f, run_sum = 0.f;

  for (int j0 = 0; j0 < 4096; j0 += 32) {
    if (j0 + 32 < 4096) {
      __builtin_prefetch(Kb + (size_t)(j0 + 32 + (lane & 15)) * ld, 0, 1);
      __builtin_prefetch(V0 + (size_t)(lane & 15) * 4096 + j0 + 32, 0, 1);
    }
    bf16x16 k0f = load_frag_row_bf(Kb + (size_t)j0 * ld, ld, lane);
    bf16x16 k1f = load_frag_row_bf(Kb + (size_t)(j0 + 16) * ld, ld, lane);
    f32x8 z0 = {}, z1 = {};
    f32x8 s0 = wmma_bf16(qf, k0f, z0);
    f32x8 s1 = wmma_bf16(qf, k1f, z1);

    // Stage scores S[16 rows][32 keys] in LDS (from C/D layout)
#pragma unroll
    for (int v = 0; v < 8; ++v) {
      S[(v + half * 8) * 32 + n]      = s0[v];
      S[(v + half * 8) * 32 + 16 + n] = s1[v];
    }
    __syncthreads();  // single-wave WG: ordering only, barrier is S_NOP

    // Online-softmax row stats (lanes 0-15 own one query row each)
    if (lane < 16) {
      const float* srow = S + lane * 32;
      float cm = -3.0e38f;
#pragma unroll
      for (int c = 0; c < 32; ++c) cm = fmaxf(cm, srow[c]);
      float nm = fmaxf(run_max, cm);
      float sc = __expf(run_max - nm);
      float cs = 0.f;
#pragma unroll
      for (int c = 0; c < 32; ++c) cs += __expf(srow[c] - nm);
      run_sum = run_sum * sc + cs;
      run_max = nm;
      stats[lane] = nm;
      stats[16 + lane] = sc;
    }
    __syncthreads();

    // Rescale running O accumulator by row factors
#pragma unroll
    for (int v = 0; v < 8; ++v) {
      const float f = stats[16 + v + half * 8];
      o0[v] *= f; o1[v] *= f;
    }

    // Build P fragment (A layout: lane row = lane&15, K = chunk-local key)
    bf16x16 p;
    {
      const float nmr = stats[n];
      const float* srow = S + n * 32;
#pragma unroll
      for (int i = 0; i < 8; ++i) {
        const int kb = ((i & 4) << 2) + half * 8 + ((i & 3) << 1);
        p[2 * i]     = (__bf16)__expf(srow[kb] - nmr);
        p[2 * i + 1] = (__bf16)__expf(srow[kb + 1] - nmr);
      }
    }

    // O += P @ V  (vT rows contiguous in keys -> two b128 loads per frag)
    bf16x16 v0f = load_frag_row_bf(V0 + j0, 4096, lane);
    bf16x16 v1f = load_frag_row_bf(V1 + j0, 4096, lane);
    o0 = wmma_bf16(p, v0f, o0);
    o1 = wmma_bf16(p, v1f, o1);
    __syncthreads();  // S reused next chunk
  }

  if (lane < 16) stats[lane] = 1.0f / run_sum;
  __syncthreads();
#pragma unroll
  for (int v = 0; v < 8; ++v) {
    const int m = v + half * 8;
    const float inv = stats[m];
    __bf16* orow = o + (size_t)(row0 + m) * 256 + h * 32;
    orow[n]      = (__bf16)(o0[v] * inv);
    orow[16 + n] = (__bf16)(o1[v] * inv);
  }
}

// ---------------------------------------------------------------------------
// Elementwise / helper kernels (float4-vectorized)
// ---------------------------------------------------------------------------
__global__ void __launch_bounds__(256)
init_kernel(const float* __restrict__ pe, const float* __restrict__ ae,
            const float* __restrict__ qe, float* __restrict__ cur,
            float* __restrict__ total) {
  const int i4 = (blockIdx.x * 256 + threadIdx.x) * 4;
  const int row = i4 >> 8, c = i4 & 255;
  float4 v;
  if (row < 3000)      v = *(const float4*)(pe + i4);
  else if (row < 3025) v = *(const float4*)(ae + (row - 3000) * 256 + c);
  else                 v = *(const float4*)(qe + (row - 3025) * 256 + c);
  *(float4*)(cur + i4) = v;
  *(float4*)(total + i4) = v;
}

__global__ void __launch_bounds__(256) zero_kernel(float* __restrict__ p) {
  *(float4*)(p + (blockIdx.x * 256 + threadIdx.x) * 4) = float4{0.f, 0.f, 0.f, 0.f};
}

// f32 -> bf16 row-major convert (4 elems/thread)
__global__ void __launch_bounds__(256)
cvt_bf16_kernel(const float* __restrict__ in, __bf16* __restrict__ out) {
  const int i4 = (blockIdx.x * 256 + threadIdx.x) * 4;
  const float4 v = *(const float4*)(in + i4);
  bf16x4 r;
  r[0] = (__bf16)v.x; r[1] = (__bf16)v.y; r[2] = (__bf16)v.z; r[3] = (__bf16)v.w;
  *(bf16x4*)(out + i4) = r;
}

// vT[c][key] = qkv[key][512 + c]; 4 keys per thread, coalesced bf16x4 writes
__global__ void __launch_bounds__(256)
vtrans_kernel(const __bf16* __restrict__ qkv, __bf16* __restrict__ vT) {
  const int i = blockIdx.x * 256 + threadIdx.x;  // < 256*1024
  const int c = i >> 10;             // dim channel 0..255
  const int j = (i & 1023) * 4;      // key
  bf16x4 r;
#pragma unroll
  for (int q = 0; q < 4; ++q) r[q] = qkv[(size_t)(j + q) * 768 + 512 + c];
  *(bf16x4*)(vT + (size_t)c * 4096 + j) = r;
}

__global__ void __launch_bounds__(256)
spmm_kernel(const int* __restrict__ rr, const int* __restrict__ cc,
            const float* __restrict__ vv, const float* __restrict__ x,
            float* __restrict__ t, int E) {
  const int idx = blockIdx.x * 256 + threadIdx.x;
  const int e = idx >> 6;
  if (e >= E) return;
  const int c4 = (idx & 63) << 2;
  const int r = rr[e], cl = cc[e];
  const float w = vv[e];
  const float4 xs = *(const float4*)(x + (size_t)cl * 256 + c4);
  float* td = t + (size_t)r * 256 + c4;
  atomicAdd(td + 0, w * xs.x);
  atomicAdd(td + 1, w * xs.y);
  atomicAdd(td + 2, w * xs.z);
  atomicAdd(td + 3, w * xs.w);
}

// out = LayerNorm(X + Y) * g + b ; optional bf16 mirror copy (one wave/row,
// lane owns 8 contiguous columns -> float4 loads/stores, bf16x8 store)
__global__ void __launch_bounds__(256)
add_ln_kernel(const float* __restrict__ X, const float* __restrict__ Y,
              const float* __restrict__ g, const float* __restrict__ b,
              float* __restrict__ out, __bf16* __restrict__ outbf, int M) {
  const int lane = threadIdx.x & 31;
  const int wave = threadIdx.x >> 5;
  const int row = blockIdx.x * 8 + wave;
  if (row >= M) return;
  const int c0 = lane * 8;
  const float* x = X + (size_t)row * 256 + c0;
  const float* y = Y + (size_t)row * 256 + c0;
  const float4 xa = *(const float4*)(x), xb = *(const float4*)(x + 4);
  const float4 ya = *(const float4*)(y), yb = *(const float4*)(y + 4);
  float v[8] = {xa.x + ya.x, xa.y + ya.y, xa.z + ya.z, xa.w + ya.w,
                xb.x + yb.x, xb.y + yb.y, xb.z + yb.z, xb.w + yb.w};
  float s = 0.f;
#pragma unroll
  for (int i = 0; i < 8; ++i) s += v[i];
#pragma unroll
  for (int off = 16; off > 0; off >>= 1) s += __shfl_xor(s, off, 32);
  const float mean = s * (1.0f / 256.0f);
  float vs = 0.f;
#pragma unroll
  for (int i = 0; i < 8; ++i) {
    const float d = v[i] - mean;
    vs += d * d;
  }
#pragma unroll
  for (int off = 16; off > 0; off >>= 1) vs += __shfl_xor(vs, off, 32);
  const float inv = rsqrtf(vs * (1.0f / 256.0f) + 1e-5f);
  const float4 ga = *(const float4*)(g + c0), gb = *(const float4*)(g + c0 + 4);
  const float4 ba = *(const float4*)(b + c0), bb = *(const float4*)(b + c0 + 4);
  const float gg[8] = {ga.x, ga.y, ga.z, ga.w, gb.x, gb.y, gb.z, gb.w};
  const float bv[8] = {ba.x, ba.y, ba.z, ba.w, bb.x, bb.y, bb.z, bb.w};
  float r[8];
#pragma unroll
  for (int i = 0; i < 8; ++i) r[i] = (v[i] - mean) * inv * gg[i] + bv[i];
  float* orow = out + (size_t)row * 256 + c0;
  *(float4*)(orow)     = float4{r[0], r[1], r[2], r[3]};
  *(float4*)(orow + 4) = float4{r[4], r[5], r[6], r[7]};
  if (outbf) {
    bf16x8 rb;
#pragma unroll
    for (int i = 0; i < 8; ++i) rb[i] = (__bf16)r[i];
    *(bf16x8*)(outbf + (size_t)row * 256 + c0) = rb;
  }
}

__global__ void __launch_bounds__(256)
accum_kernel(const float* __restrict__ x2, float* __restrict__ cur,
             float* __restrict__ total) {
  const int i4 = (blockIdx.x * 256 + threadIdx.x) * 4;
  const float4 a = *(const float4*)(cur + i4);
  const float4 d = *(const float4*)(x2 + i4);
  const float4 c = float4{a.x + d.x, a.y + d.y, a.z + d.z, a.w + d.w};
  const float4 tt = *(const float4*)(total + i4);
  *(float4*)(cur + i4) = c;
  *(float4*)(total + i4) = float4{tt.x + c.x, tt.y + c.y, tt.z + c.z, tt.w + c.w};
}

__global__ void __launch_bounds__(256)
copyout_kernel(const float* __restrict__ total, float* __restrict__ out, int n) {
  const int i4 = (blockIdx.x * 256 + threadIdx.x) * 4;
  const float4 v = *(const float4*)(total + i4);
  *(float4*)(out + i4) = v;
  *(float4*)(out + i4 + n) = v;
}

// ---------------------------------------------------------------------------
// Launcher
// ---------------------------------------------------------------------------
extern "C" void kernel_launch(void* const* d_in, const int* in_sizes, int n_in,
                              void* d_out, int out_size, void* d_ws, size_t ws_size,
                              hipStream_t stream) {
  const int*   adj_row = (const int*)d_in[0];
  const int*   adj_col = (const int*)d_in[1];
  const float* adj_val = (const float*)d_in[2];
  const float* pe      = (const float*)d_in[3];
  const float* ae      = (const float*)d_in[4];
  const float* qe      = (const float*)d_in[5];
  const float* Wqkv    = (const float*)d_in[6];
  const float* Wo      = (const float*)d_in[7];
  const float* W1      = (const float*)d_in[8];
  const float* b1      = (const float*)d_in[9];
  const float* W2      = (const float*)d_in[10];
  const float* b2      = (const float*)d_in[11];
  const float* ln1_g   = (const float*)d_in[12];
  const float* ln1_b   = (const float*)d_in[13];
  const float* ln2_g   = (const float*)d_in[14];
  const float* ln2_b   = (const float*)d_in[15];

  const int    N  = 4096, E = 131072;
  const size_t NM = (size_t)N * 256;  // 1048576

  float* ws    = (float*)d_ws;
  float*  cur    = ws;
  float*  total  = ws + 1 * NM;
  float*  t      = ws + 2 * NM;             // h2 aliases t
  float*  x1     = ws + 3 * NM;
  float*  aproj  = ws + 4 * NM;             // x2 aliases aproj
  __bf16* h1bf   = (__bf16*)(ws + 5 * NM);  // 4*NM bf16
  __bf16* qkvbf  = (__bf16*)(ws + 7 * NM);  // 3*NM bf16
  __bf16* attnbf = (__bf16*)(ws + 8 * NM + NM / 2);
  __bf16* vT     = (__bf16*)(ws + 9 * NM);
  __bf16* tbf    = (__bf16*)(ws + 9 * NM + NM / 2);
  __bf16* x1bf   = (__bf16*)(ws + 10 * NM);
  __bf16* wq_p   = (__bf16*)(ws + 10 * NM + NM / 2);
  __bf16* wo_p   = wq_p + 196608;   // 256x768 packed
  __bf16* w1_p   = wo_p + 65536;    // 256x256 packed
  __bf16* w2_p   = w1_p + 262144;   // 256x1024 packed
  float*  h2     = t;
  float*  x2     = aproj;

  const int eb  = (int)(NM / 256);
  const int eb4 = eb / 4;
  const float qscale = 0.17677669529663687f;  // 1/sqrt(32)

  // Pack weights to bf16 fragment layout; fold 1/sqrt(DH) into Q columns.
  pack_w_kernel<<<48, 256, 0, stream>>>(Wqkv, wq_p, 256, 768, 256, qscale);
  pack_w_kernel<<<16, 256, 0, stream>>>(Wo,   wo_p, 256, 256, 0, 1.0f);
  pack_w_kernel<<<64, 256, 0, stream>>>(W1,   w1_p, 256, 1024, 0, 1.0f);
  pack_w_kernel<<<64, 256, 0, stream>>>(W2,   w2_p, 1024, 256, 0, 1.0f);

  init_kernel<<<eb4, 256, 0, stream>>>(pe, ae, qe, cur, total);
  for (int blk = 0; blk < 3; ++blk) {
    zero_kernel<<<eb4, 256, 0, stream>>>(t);
    spmm_kernel<<<(E * 64) / 256, 256, 0, stream>>>(adj_row, adj_col, adj_val,
                                                    cur, t, E);
    cvt_bf16_kernel<<<eb4, 256, 0, stream>>>(t, tbf);
    // qkv = t @ Wqkv (bf16 out; Q columns pre-scaled)
    gemm_wmma<false, false, true><<<192, 256, 0, stream>>>(
        tbf, wq_p, nullptr, qkvbf, N, 256, 768);
    vtrans_kernel<<<1024, 256, 0, stream>>>(qkvbf, vT);
    attention_kernel<<<2048, 32, 0, stream>>>(qkvbf, vT, attnbf);
    // aproj = attn @ Wo (f32 out)
    gemm_wmma<false, false, false><<<64, 256, 0, stream>>>(
        attnbf, wo_p, nullptr, aproj, N, 256, 256);
    add_ln_kernel<<<N / 8, 256, 0, stream>>>(t, aproj, ln1_g, ln1_b, x1, x1bf, N);
    // h1 = relu(x1 @ W1 + b1) (bf16 out)
    gemm_wmma<true, true, true><<<256, 256, 0, stream>>>(
        x1bf, w1_p, b1, h1bf, N, 256, 1024);
    // h2 = h1 @ W2 + b2 (f32 out)
    gemm_wmma<false, true, false><<<64, 256, 0, stream>>>(
        h1bf, w2_p, b2, h2, N, 1024, 256);
    add_ln_kernel<<<N / 8, 256, 0, stream>>>(x1, h2, ln2_g, ln2_b, x2, nullptr, N);
    accum_kernel<<<eb4, 256, 0, stream>>>(x2, cur, total);
  }
  copyout_kernel<<<eb4, 256, 0, stream>>>(total, (float*)d_out, (int)NM);
}
